// MonarchAttention_84396107366847
// MI455X (gfx1250) — matrix operand order repaired and checked by reading, not compiled, not run on hardware
//
#include <hip/hip_runtime.h>

#define NH   12
#define HD   64
#define SEQ  2912
#define B1C  28
#define B2C  52
#define FC   2
#define FKC  56          // F*B1
#define FJ   104         // F*B2
#define BB   1456        // B1*B2
#define SCL  0.125f
#define NTILE 91         // 2912/32

typedef __attribute__((ext_vector_type(16))) _Float16 v16h;
typedef __attribute__((ext_vector_type(8)))  _Float16 v8h;
typedef __attribute__((ext_vector_type(4)))  _Float16 v4h;
typedef __attribute__((ext_vector_type(8)))  float    v8f;
typedef __attribute__((ext_vector_type(4)))  float    v4f;

// ---------------- RoPE only: y[s,h,:] = rope(x[s,h,:], cos[s], sin[s]) ----------------
__global__ void k_rope(const float* __restrict__ x, const float* __restrict__ cosv,
                       const float* __restrict__ sinv, float* __restrict__ y) {
  int t = blockIdx.x * 256 + threadIdx.x;             // pair index
  if (t >= SEQ * NH * 32) return;
  int m = t & 31, sh = t >> 5, s = sh / NH;
  float c = cosv[s * 32 + m], sn = sinv[s * 32 + m];
  float x1 = x[(size_t)sh * 64 + 2 * m], x2 = x[(size_t)sh * 64 + 2 * m + 1];
  y[(size_t)sh * 64 + 2 * m]     = x1 * c - x2 * sn;
  y[(size_t)sh * 64 + 2 * m + 1] = x2 * c + x1 * sn;
}

// ------ blin (+optional rope), tiled: one block = 16 seq positions x 1 head -----------
__global__ void __launch_bounds__(256)
k_blin16(const float* __restrict__ x, const float* __restrict__ W,
         const float* __restrict__ b, const float* __restrict__ cosv,
         const float* __restrict__ sinv, float* __restrict__ y, int do_rope) {
  int s0 = blockIdx.x * 16, h = blockIdx.y, tid = threadIdx.x;
  __shared__ float Wl[64 * 64];     // W[h][e][d]
  __shared__ float xs[16 * 64];
  __shared__ float ys[16 * 64];
  for (int u = tid; u < 4096; u += 256) Wl[u] = W[(size_t)h * 4096 + u];
  for (int u = tid; u < 1024; u += 256) {
    int si = u >> 6, d = u & 63;
    xs[u] = x[((size_t)(s0 + si) * NH + h) * 64 + d];
  }
  __syncthreads();
  for (int u = tid; u < 1024; u += 256) {
    int si = u >> 6, e = u & 63;
    float acc = b[h * 64 + e];
    const float* w = &Wl[e * 64];
    const float* xr = &xs[si * 64];
    for (int d = 0; d < 64; ++d) acc += xr[d] * w[d];
    ys[u] = acc;
  }
  __syncthreads();
  for (int u = tid; u < 1024; u += 256) {
    int si = u >> 6, e = u & 63;
    float o = ys[u];
    if (do_rope) {
      int m = e >> 1, s = s0 + si;
      float c = cosv[s * 32 + m], sn = sinv[s * 32 + m];
      o = (e & 1) ? (ys[si * 64 + e] * c + ys[si * 64 + e - 1] * sn)
                  : (ys[si * 64 + e] * c - ys[si * 64 + e + 1] * sn);
    }
    y[((size_t)(s0 + si) * NH + h) * 64 + e] = o;
  }
}

// ------------- slice + blin + rope (lkq / rqq); g=(f,r), s0 = f*BB + r*strideMul -------
__global__ void k_slice(const float* __restrict__ x, const float* __restrict__ W,
                        const float* __restrict__ b, const float* __restrict__ cosv,
                        const float* __restrict__ sinv, float* __restrict__ y,
                        int G1, int strideMul) {
  int g = blockIdx.x, h = blockIdx.y, e = threadIdx.x;
  int f = g / G1, r = g - f * G1;
  int s0 = f * BB + r * strideMul;
  __shared__ float xs[64], ys[64];
  xs[e] = x[((size_t)s0 * NH + h) * 64 + e];
  __syncthreads();
  const float* w = W + ((size_t)h * 64 + e) * 64;
  float acc = b[h * 64 + e];
  for (int d = 0; d < 64; ++d) acc += xs[d] * w[d];
  ys[e] = acc;
  __syncthreads();
  int m = e >> 1;
  float c = cosv[s0 * 32 + m], sn = sinv[s0 * 32 + m];
  float o = (e & 1) ? (ys[e] * c + ys[e - 1] * sn) : (ys[e] * c - ys[e + 1] * sn);
  y[((size_t)g * NH + h) * 64 + e] = o;
}

// --------- tiny SDPA (1 query, Nk keys at s = base + jj*step) + output rope ------------
__global__ void k_sdpa(const float* __restrict__ q, const float* __restrict__ K,
                       const float* __restrict__ V, const float* __restrict__ cosv,
                       const float* __restrict__ sinv, float* __restrict__ out,
                       int G1, int baseMul, int step, int Nk) {
  int g = blockIdx.x, h = blockIdx.y, e = threadIdx.x;
  int f = g / G1, r = g - f * G1;
  int base = f * BB + r * baseMul;
  __shared__ float qs[64], ps[64], ys[64];
  qs[e] = q[((size_t)g * NH + h) * 64 + e];
  __syncthreads();
  float pe = -1e30f;
  if (e < Nk) {
    const float* kp = K + ((size_t)(base + e * step) * NH + h) * 64;
    float d0 = 0.f;
    for (int d = 0; d < 64; ++d) d0 += qs[d] * kp[d];
    pe = d0 * SCL;
  }
  ps[e] = pe;
  __syncthreads();
  float mx = -1e30f;
  for (int jj = 0; jj < Nk; ++jj) mx = fmaxf(mx, ps[jj]);
  __syncthreads();
  ps[e] = (e < Nk) ? __expf(pe - mx) : 0.f;
  __syncthreads();
  float acc = 0.f, ell = 0.f;
  for (int jj = 0; jj < Nk; ++jj) {
    float p = ps[jj]; ell += p;
    acc += p * V[((size_t)(base + jj * step) * NH + h) * 64 + e];
  }
  ys[e] = acc / ell;
  __syncthreads();
  int m = e >> 1;
  float c = cosv[base * 32 + m], sn = sinv[base * 32 + m];
  float o = (e & 1) ? (ys[e] * c + ys[e - 1] * sn) : (ys[e] * c - ys[e + 1] * sn);
  out[((size_t)g * NH + h) * 64 + e] = o;
}

// ---- Lg[h][a][j][i][fk] = sum_d lq_r[a,i,j,h,d] * lk_out[fk,h,d]; grid (j,a,h) --------
__global__ void __launch_bounds__(256)
k_Lg(const float* __restrict__ lq_r, const float* __restrict__ lk_out,
     float* __restrict__ Lg) {
  int j = blockIdx.x, a = blockIdx.y, h = blockIdx.z, tid = threadIdx.x;
  __shared__ float Qs[B1C * 64];
  __shared__ float Ks[FKC * 64];
  for (int u = tid; u < B1C * 64; u += 256) {
    int i = u >> 6, d = u & 63;
    Qs[u] = lq_r[((size_t)(a * BB + i * B2C + j) * NH + h) * 64 + d];
  }
  for (int u = tid; u < FKC * 64; u += 256) {
    int fk = u >> 6, d = u & 63;
    Ks[u] = lk_out[((size_t)fk * NH + h) * 64 + d];
  }
  __syncthreads();
  for (int u = tid; u < B1C * FKC; u += 256) {
    int i = u / FKC, fk = u - i * FKC;
    float acc = 0.f;
    for (int d = 0; d < 64; ++d) acc += Qs[i * 64 + d] * Ks[fk * 64 + d];
    Lg[(((size_t)(h * 2 + a) * B2C + j) * B1C + i) * FKC + fk] = acc;
  }
}

// ---- Rg[h][a][j][fk][l] = sum_d rq_out[a*52+j,h,d] * rk_r[f*BB+k*52+l,h,d]; grid (fk2,a,h)
__global__ void __launch_bounds__(256)
k_Rg(const float* __restrict__ rq_out, const float* __restrict__ rk_r,
     float* __restrict__ Rg) {
  int fk2 = blockIdx.x, a = blockIdx.y, h = blockIdx.z, tid = threadIdx.x;
  __shared__ float Qs[B2C * 64];
  __shared__ float Ks[2 * B2C * 64];
  for (int u = tid; u < B2C * 64; u += 256) {
    int jq = u >> 6, d = u & 63;
    Qs[u] = rq_out[((size_t)(a * B2C + jq) * NH + h) * 64 + d];
  }
  for (int u = tid; u < 2 * B2C * 64; u += 256) {
    int kk = u >> 6, d = u & 63;
    int fk = fk2 * 2 + kk / B2C, l = kk % B2C;
    int f = fk / B1C, k = fk - f * B1C;
    int s = f * BB + k * B2C + l;
    Ks[u] = rk_r[((size_t)s * NH + h) * 64 + d];
  }
  __syncthreads();
  for (int u = tid; u < B2C * 2 * B2C; u += 256) {
    int jq = u / (2 * B2C), kk = u - jq * 2 * B2C;
    float acc = 0.f;
    for (int d = 0; d < 64; ++d) acc += Qs[jq * 64 + d] * Ks[kk * 64 + d];
    int fk = fk2 * 2 + kk / B2C, l = kk % B2C;
    Rg[(((size_t)(h * 2 + a) * B2C + jq) * FKC + fk) * B2C + l] = acc;
  }
}

// ---------------- v -> f16, layout [h][d][t] ----------------
__global__ void k_vcvt(const float* __restrict__ v, _Float16* __restrict__ v16t) {
  int idx = blockIdx.x * 256 + threadIdx.x;
  if (idx >= NH * HD * SEQ) return;
  int t = idx % SEQ, hd = idx / SEQ;
  int h = hd >> 6, d = hd & 63;
  v16t[idx] = (_Float16)v[((size_t)t * NH + h) * HD + d];
}

// ---------------- fused Monarch attention; grid (j, a, h), 256 threads -----------------
__global__ void __launch_bounds__(256)
k_monarch(const float* __restrict__ Lg, const float* __restrict__ Rg,
          const _Float16* __restrict__ v16t, float* __restrict__ out) {
  const int j = blockIdx.x, a = blockIdx.y, h = blockIdx.z;
  const int tid = threadIdx.x;

  __shared__ float Lld[B1C * FKC];                       // [i][fk]
  __shared__ __align__(16) float Rld[FKC * B2C];         // linear in t = fk*52+l
  __shared__ __align__(32) _Float16 Pbuf[2 * 1024];      // double-buffered 32x32 P tile
  __shared__ float red[256];
  __shared__ float m_i[32], el_i[32];

  const float* Lb = Lg + (((size_t)(h * 2 + a) * B2C + j) * B1C) * FKC;
  for (int u = tid; u < B1C * FKC; u += 256) Lld[u] = Lb[u];
  const float* Rb = Rg + (((size_t)(h * 2 + a) * B2C + j) * FKC) * B2C;
  for (int u = tid; u < FKC * B2C; u += 256) Rld[u] = Rb[u];
  __syncthreads();

  // thread -> (row i, 4-key group kq); groups of 4 keys never span >1 fk boundary
  const int qi = tid >> 3, sub = tid & 7;
  const int irow = (qi < B1C) ? qi : 0;
  const float gate = (qi < B1C) ? 1.f : 0.f;
  const float* Lrow = &Lld[irow * FKC];

  // ---- pass 1: row max of L[i,fk]*R[t] (no exp) ----
  float mx = -1e30f;
  for (int n = 0; n < NTILE; ++n) {
    int t0 = (n * 8 + sub) * 4;
    unsigned fk0 = (unsigned)t0 / 52u;
    int l0 = t0 - (int)fk0 * 52;
    int fk1 = (int)fk0 + ((l0 >= 49) ? 1 : 0);
    float lv0 = Lrow[fk0], lv1 = Lrow[fk1];
    v4f rr = *(const v4f*)&Rld[t0];
    float p0 = rr[0] * ((l0 + 0 < 52) ? lv0 : lv1);
    float p1 = rr[1] * ((l0 + 1 < 52) ? lv0 : lv1);
    float p2 = rr[2] * ((l0 + 2 < 52) ? lv0 : lv1);
    float p3 = rr[3] * ((l0 + 3 < 52) ? lv0 : lv1);
    mx = fmaxf(mx, fmaxf(fmaxf(p0, p1), fmaxf(p2, p3)));
  }
  red[tid] = mx;
  __syncthreads();
  if (sub == 0 && qi < B1C) {
    float m = red[tid];
    for (int r = 1; r < 8; ++r) m = fmaxf(m, red[tid + r]);
    m_i[qi] = m * SCL;
  }
  __syncthreads();
  const float mreg = (qi < B1C) ? m_i[qi] : 0.f;

  // ---- main loop: software-pipelined P'(f16) tile build + WMMA P'.V ----
  const int wave = tid >> 5, lane = tid & 31;
  const int mt = wave >> 2, nt = wave & 3;
  const int ln = lane & 15, hh = lane >> 4;
  v8f acc = {0.f, 0.f, 0.f, 0.f, 0.f, 0.f, 0.f, 0.f};
  const _Float16* vbase = v16t + ((size_t)(h * HD + nt * 16 + ln)) * SEQ + hh * 16;
  const int apOff = (mt * 16 + ln) * 32 + hh * 16;
  float sumP = 0.f;

  auto computeP = [&](int tile) {
    const int buf = (tile & 1) * 1024;
    int t0 = tile * 32 + sub * 4;
    unsigned fk0 = (unsigned)t0 / 52u;
    int l0 = t0 - (int)fk0 * 52;
    int fk1 = (int)fk0 + ((l0 >= 49) ? 1 : 0);
    float lv0 = gate * Lrow[fk0], lv1 = gate * Lrow[fk1];
    v4f rr = *(const v4f*)&Rld[t0];
    float p0 = gate * __expf(fmaf(rr[0] * ((l0 + 0 < 52) ? lv0 : lv1), SCL, -mreg));
    float p1 = gate * __expf(fmaf(rr[1] * ((l0 + 1 < 52) ? lv0 : lv1), SCL, -mreg));
    float p2 = gate * __expf(fmaf(rr[2] * ((l0 + 2 < 52) ? lv0 : lv1), SCL, -mreg));
    float p3 = gate * __expf(fmaf(rr[3] * ((l0 + 3 < 52) ? lv0 : lv1), SCL, -mreg));
    sumP += (p0 + p1) + (p2 + p3);
    v4h hp; hp[0] = (_Float16)p0; hp[1] = (_Float16)p1;
    hp[2] = (_Float16)p2; hp[3] = (_Float16)p3;
    *(v4h*)&Pbuf[buf + qi * 32 + sub * 4] = hp;
  };

  union AF { v16h v; v8h p[2]; };
  AF af;
  // prologue: tile 0 P in buf0, then prefetch its fragments
  computeP(0);
  __syncthreads();
  af.p[0] = *(const v8h*)&Pbuf[apOff];
  af.p[1] = *(const v8h*)&Pbuf[apOff + 8];
  v16h bf = *(const v16h*)vbase;

  for (int tile = 0; tile < NTILE; ++tile) {
    // build next P tile while afrag/bfrag loads for this tile are in flight
    if (tile + 1 < NTILE) computeP(tile + 1);
    acc = __builtin_amdgcn_wmma_f32_16x16x32_f16(false, af.v, false, bf,
                                                 (short)0, acc, false, false);
    __syncthreads();     // next P tile visible; prev A-frag reads already complete
    if (tile + 1 < NTILE) {
      const int nb = ((tile + 1) & 1) * 1024;
      af.p[0] = *(const v8h*)&Pbuf[nb + apOff];
      af.p[1] = *(const v8h*)&Pbuf[nb + apOff + 8];
      bf = *(const v16h*)(vbase + (tile + 1) * 32);
    }
  }

  // ---- row sums -> 1/ell, then scaled output write ----
  red[tid] = sumP;
  __syncthreads();
  if (sub == 0 && qi < B1C) {
    float s = 0.f;
    for (int r = 0; r < 8; ++r) s += red[tid + r];
    el_i[qi] = __builtin_amdgcn_rcpf(s);
  }
  __syncthreads();

  // D layout: element r -> M = hh*8 + r (within tile), N = ln
#pragma unroll
  for (int r = 0; r < 8; ++r) {
    int iq = mt * 16 + hh * 8 + r;
    if (iq < B1C) {
      int s = a * BB + iq * B2C + j;
      out[((size_t)s * NH + h) * HD + nt * 16 + ln] = acc[r] * el_i[iq];
    }
  }
}

extern "C" void kernel_launch(void* const* d_in, const int* in_sizes, int n_in,
                              void* d_out, int out_size, void* d_ws, size_t ws_size,
                              hipStream_t stream) {
  (void)in_sizes; (void)n_in; (void)out_size; (void)ws_size;
  const float* lq   = (const float*)d_in[0];
  const float* lk   = (const float*)d_in[1];
  const float* rq   = (const float*)d_in[2];
  const float* rk   = (const float*)d_in[3];
  const float* v    = (const float*)d_in[4];
  const float* cosv = (const float*)d_in[5];
  const float* sinv = (const float*)d_in[6];
  const float* W_lkq = (const float*)d_in[7];  const float* b_lkq = (const float*)d_in[8];
  const float* W_lkk = (const float*)d_in[9];  const float* b_lkk = (const float*)d_in[10];
  const float* W_lkv = (const float*)d_in[11]; const float* b_lkv = (const float*)d_in[12];
  const float* W_rqq = (const float*)d_in[13]; const float* b_rqq = (const float*)d_in[14];
  const float* W_rqk = (const float*)d_in[15]; const float* b_rqk = (const float*)d_in[16];
  const float* W_rqv = (const float*)d_in[17]; const float* b_rqv = (const float*)d_in[18];
  float* out = (float*)d_out;

  float* ws = (float*)d_ws;
  size_t off = 0;
  auto alloc = [&](size_t n) { float* p = ws + off; off += (n + 7) & ~(size_t)7; return p; };
  const size_t SHD = (size_t)SEQ * NH * HD;
  float* lq_r   = alloc(SHD);
  float* rk_r   = alloc(SHD);
  float* lkk    = alloc(SHD);
  float* lkv    = alloc(SHD);
  float* rqk    = alloc(SHD);
  float* rqv    = alloc(SHD);
  float* lkq    = alloc((size_t)FKC * NH * HD);
  float* rqq    = alloc((size_t)FJ * NH * HD);
  float* lk_out = alloc((size_t)FKC * NH * HD);
  float* rq_out = alloc((size_t)FJ * NH * HD);
  float* Lg     = alloc((size_t)NH * FC * B2C * B1C * FKC);
  float* Rg     = alloc((size_t)NH * FC * B2C * FKC * B2C);
  _Float16* v16t = (_Float16*)(ws + off);

  dim3 b64(64), b256(256);
  int ropeBlocks = (SEQ * NH * 32 + 255) / 256;
  k_rope<<<dim3(ropeBlocks), b256, 0, stream>>>(lq, cosv, sinv, lq_r);
  k_rope<<<dim3(ropeBlocks), b256, 0, stream>>>(rk, cosv, sinv, rk_r);
  k_blin16<<<dim3(SEQ / 16, NH), b256, 0, stream>>>(lk, W_lkk, b_lkk, cosv, sinv, lkk, 1);
  k_blin16<<<dim3(SEQ / 16, NH), b256, 0, stream>>>(lk, W_lkv, b_lkv, cosv, sinv, lkv, 0);
  k_blin16<<<dim3(SEQ / 16, NH), b256, 0, stream>>>(rq, W_rqk, b_rqk, cosv, sinv, rqk, 1);
  k_blin16<<<dim3(SEQ / 16, NH), b256, 0, stream>>>(rq, W_rqv, b_rqv, cosv, sinv, rqv, 0);
  k_slice<<<dim3(FKC, NH), b64, 0, stream>>>(lk, W_lkq, b_lkq, cosv, sinv, lkq, B1C, B2C);
  k_slice<<<dim3(FJ,  NH), b64, 0, stream>>>(rq, W_rqq, b_rqq, cosv, sinv, rqq, B2C, 1);
  k_sdpa<<<dim3(FKC, NH), b64, 0, stream>>>(lkq, lkk, lkv, cosv, sinv, lk_out,
                                            B1C, B2C, 1, B2C);
  k_sdpa<<<dim3(FJ,  NH), b64, 0, stream>>>(rqq, rqk, rqv, cosv, sinv, rq_out,
                                            B2C, 1, B2C, B1C);
  k_Lg<<<dim3(B2C, FC, NH), b256, 0, stream>>>(lq_r, lk_out, Lg);
  k_Rg<<<dim3(B1C, FC, NH), b256, 0, stream>>>(rq_out, rk_r, Rg);
  k_vcvt<<<dim3((NH * HD * SEQ + 255) / 256), b256, 0, stream>>>(v, v16t);
  k_monarch<<<dim3(B2C, FC, NH), b256, 0, stream>>>(Lg, Rg, v16t, out);
}
